// PlainGCN_47991964565664
// MI455X (gfx1250) — matrix operand order repaired and checked by reading, not compiled, not run on hardware
//
#include <hip/hip_runtime.h>
#include <math.h>

typedef __attribute__((ext_vector_type(16))) _Float16 v16h;
typedef __attribute__((ext_vector_type(8)))  _Float16 v8h;
typedef __attribute__((ext_vector_type(8)))  float    v8f;

constexpr int   BATCH  = 8;
constexpr int   P      = 8192;
constexpr int   NPTS   = BATCH * P;   // 65536
constexpr int   KNN_K  = 16;
constexpr float EPSV   = 1e-5f;
constexpr int   PPW    = 4;           // points per wave in edge kernels

// Wave-local LDS ordering fence: same-wave DS ops are in-order on CDNA5, so a
// dscnt wait + compiler barrier is enough for intra-wave producer->consumer
// through LDS (also blocks TBAA-based reordering across our type-punned LDS).
__device__ __forceinline__ void wave_lds_fence() {
  asm volatile("s_wait_dscnt 0" ::: "memory");
}

__device__ __forceinline__ void st2h(_Float16* p, float a, float b) {
  union { _Float16 h[2]; unsigned u; } t;
  t.h[0] = (_Float16)a; t.h[1] = (_Float16)b;
  *(unsigned*)p = t.u;
}

// ---------------------------------------------------------------------------
// KNN: brute force per batch, LDS float4 candidate tiles, per-lane register
// top-16 insertion sort, wave32 shuffle argmin merge. One wave per query row.
// ---------------------------------------------------------------------------
constexpr int KNN_CHUNK = 2048;   // float4 tile = 32 KB LDS

__global__ __launch_bounds__(256) void knn_kernel(const float* __restrict__ vc,
                                                  int* __restrict__ knn_out) {
  __shared__ float4 sp[KNN_CHUNK];
  const float4* vcv = (const float4*)vc;
  const int lane    = threadIdx.x & 31;
  const int wv      = threadIdx.x >> 5;
  const int b       = blockIdx.x >> 7;          // 128 blocks per batch
  const int rowbase = (blockIdx.x & 127) * 64;  // 64 rows per block
  const int boff    = b * P;

  for (int rg = 0; rg < 8; ++rg) {
    const int    row = rowbase + wv * 8 + rg;
    const float4 rp  = vcv[boff + row];
    const float  px = rp.y, py = rp.z, pz = rp.w;

    float dbest[16];
    int   nbest[16];
#pragma unroll
    for (int t = 0; t < 16; ++t) { dbest[t] = 3.4e38f; nbest[t] = row; }

    for (int ch = 0; ch < P; ch += KNN_CHUNK) {
      __syncthreads();
      for (int t = threadIdx.x; t < KNN_CHUNK; t += 256)
        sp[t] = vcv[boff + ch + t];
      __syncthreads();
      if (rg == 0 && ch + KNN_CHUNK < P) {  // warm L2/L0 for the next tile
        for (int t = threadIdx.x; t < KNN_CHUNK; t += 256)
          __builtin_prefetch((const void*)(vcv + boff + ch + KNN_CHUNK + t), 0, 1);
      }
#pragma unroll 2
      for (int j = lane; j < KNN_CHUNK; j += 32) {
        float4 q  = sp[j];
        float  dx = q.y - px, dy = q.z - py, dz = q.w - pz;
        float  d  = dx * dx + dy * dy + dz * dz;
        if (d < dbest[15]) {
          dbest[15] = d; nbest[15] = ch + j;
#pragma unroll
          for (int t = 15; t > 0; --t) {
            if (dbest[t] < dbest[t - 1]) {
              float td = dbest[t]; dbest[t] = dbest[t - 1]; dbest[t - 1] = td;
              int   ti = nbest[t]; nbest[t] = nbest[t - 1]; nbest[t - 1] = ti;
            }
          }
        }
      }
    }
    // merge: 16 rounds of wave argmin over each lane's sorted head
    for (int round = 0; round < KNN_K; ++round) {
      float bd = dbest[0];
      int   bl = lane;
#pragma unroll
      for (int m = 16; m >= 1; m >>= 1) {
        float od = __shfl_xor(bd, m, 32);
        int   ol = __shfl_xor(bl, m, 32);
        if (od < bd || (od == bd && ol < bl)) { bd = od; bl = ol; }
      }
      if (lane == bl) {
        knn_out[(size_t)(boff + row) * KNN_K + round] = boff + nbest[0];
#pragma unroll
        for (int t = 0; t < 15; ++t) { dbest[t] = dbest[t + 1]; nbest[t] = nbest[t + 1]; }
        dbest[15] = 3.4e38f;
      }
    }
  }
}

// ---------------------------------------------------------------------------
// Shared edge helpers: E is staged in LDS transposed, E_T[neighbor][Kdim],
// so B fragments are two contiguous ds_load_b128 per K-tile.
// ---------------------------------------------------------------------------
template <int CIN, int COUT, int KPAD>
__device__ __forceinline__ void build_eT(_Float16* e, const float* __restrict__ xin,
                                         int n, int j, int k, int half) {
  constexpr int CH = CIN / 2;   // channels per lane-half (even)
  const int cbase = half * CH;
#pragma unroll
  for (int t = 0; t < CH; t += 2) {
    const int c = cbase + t;
    float2 xi2 = *(const float2*)(xin + (size_t)n * CIN + c);
    float2 xj2 = *(const float2*)(xin + (size_t)j * CIN + c);
    st2h(e + k * KPAD + c,        xi2.x,         xi2.y);
    st2h(e + k * KPAD + CIN + c,  xj2.x - xi2.x, xj2.y - xi2.y);
  }
#pragma unroll
  for (int rr = 2 * CIN + 2 * half; rr < KPAD; rr += 4)
    *(unsigned*)(e + k * KPAD + rr) = 0u;
}

template <int KPAD>
__device__ __forceinline__ void load_bfr(v16h* bfr, const _Float16* e, int lane) {
  constexpr int KT = KPAD / 32;
  const int N = lane & 15;
#pragma unroll
  for (int kt = 0; kt < KT; ++kt) {
    const _Float16* bp = e + (size_t)N * KPAD + kt * 32 + ((lane < 16) ? 0 : 8);
    v8h lo = *(const v8h*)bp;
    v8h hi = *(const v8h*)(bp + 16);
    bfr[kt] = __builtin_shufflevector(lo, hi, 0, 1, 2, 3, 4, 5, 6, 7,
                                      8, 9, 10, 11, 12, 13, 14, 15);
  }
}

// ---------------------------------------------------------------------------
// EdgeConv pass A (stats): h = relu(W @ e + b) via v_wmma_f32_16x16x32_f16.
// Per-channel sum / sumsq accumulated in REGISTERS across the point loop,
// reduced once at kernel end (shfl -> LDS atomics -> global atomics).
// ---------------------------------------------------------------------------
template <int CIN, int COUT, int KPAD>
__global__ __launch_bounds__(256) void edge_stats_kernel(
    const float* __restrict__ xin, const int* __restrict__ knn_in,
    const float* __restrict__ Wg, const float* __restrict__ bg,
    float* __restrict__ stats) {
  constexpr int MT = COUT / 16, KT = KPAD / 32;
  __shared__ _Float16 sW[COUT * KPAD];
  __shared__ float    sB[COUT];
  __shared__ _Float16 sE[8][16 * KPAD];   // per-wave E_T[neighbor][Kdim]
  __shared__ float    sSum[COUT], sSq[COUT];

  const int tid = threadIdx.x, lane = tid & 31, wv = tid >> 5;
  for (int t = tid; t < COUT * KPAD; t += 256) {
    int o = t / KPAD, kk = t % KPAD;
    sW[t] = (kk < 2 * CIN) ? (_Float16)Wg[o * 2 * CIN + kk] : (_Float16)0.f;
  }
  for (int t = tid; t < COUT; t += 256) { sB[t] = bg[t]; sSum[t] = 0.f; sSq[t] = 0.f; }
  __syncthreads();

  // A fragments (weights), hoisted across the point loop (16-bit A layout)
  v16h afr[MT][KT];
  {
    const int Mrow = lane & 15;
    const int kof  = (lane < 16) ? 0 : 8;
#pragma unroll
    for (int mt = 0; mt < MT; ++mt)
#pragma unroll
      for (int kt = 0; kt < KT; ++kt)
#pragma unroll
        for (int v = 0; v < 8; ++v) {
          int kk = kt * 32 + (v >> 2) * 16 + kof + 2 * (v & 3);
          afr[mt][kt][2 * v]     = sW[(mt * 16 + Mrow) * KPAD + kk];
          afr[mt][kt][2 * v + 1] = sW[(mt * 16 + Mrow) * KPAD + kk + 1];
        }
  }

  _Float16* e    = sE[wv];
  const int k    = lane & 15;
  const int half = lane >> 4;
  const int n0   = (blockIdx.x * 8 + wv) * PPW;

  float accS[MT][8], accQ[MT][8];
#pragma unroll
  for (int mt = 0; mt < MT; ++mt)
#pragma unroll
    for (int r = 0; r < 8; ++r) { accS[mt][r] = 0.f; accQ[mt][r] = 0.f; }

  for (int pi = 0; pi < PPW; ++pi) {
    const int n = n0 + pi;
    wave_lds_fence();                       // prior B reads done before E rewrite
    const int j = knn_in[(size_t)n * KNN_K + k];
    build_eT<CIN, COUT, KPAD>(e, xin, n, j, k, half);
    wave_lds_fence();                       // E visible to all lanes of the wave

    v16h bfr[KT];
    load_bfr<KPAD>(bfr, e, lane);

#pragma unroll
    for (int mt = 0; mt < MT; ++mt) {
      v8f cf = {};
#pragma unroll
      for (int kt = 0; kt < KT; ++kt)
        cf = __builtin_amdgcn_wmma_f32_16x16x32_f16(false, afr[mt][kt], false, bfr[kt],
                                                    (short)0, cf, false, false);
#pragma unroll
      for (int r = 0; r < 8; ++r) {
        int   M   = mt * 16 + r + ((lane < 16) ? 0 : 8);
        float val = cf[r] + sB[M];
        val       = val > 0.f ? val : 0.f;
        accS[mt][r] += val;
        accQ[mt][r] += val * val;
      }
    }
  }

  // single cross-lane reduction at the end (over the 16 lanes sharing M)
#pragma unroll
  for (int mt = 0; mt < MT; ++mt)
#pragma unroll
    for (int r = 0; r < 8; ++r) {
      float s = accS[mt][r], q = accQ[mt][r];
#pragma unroll
      for (int m = 1; m <= 8; m <<= 1) { s += __shfl_xor(s, m, 32); q += __shfl_xor(q, m, 32); }
      if ((lane & 15) == 0) {
        int M = mt * 16 + r + ((lane < 16) ? 0 : 8);
        atomicAdd(&sSum[M], s);
        atomicAdd(&sSq[M], q);
      }
    }
  __syncthreads();
  for (int t = tid; t < COUT; t += 256) {
    atomicAdd(&stats[t], sSum[t]);
    atomicAdd(&stats[COUT + t], sSq[t]);
  }
}

// ---------------------------------------------------------------------------
// Finalize: (sum, sumsq) -> fused affine (a, b): hn = h*a + b
// ---------------------------------------------------------------------------
__global__ void finalize_kernel(const float* __restrict__ stats,
                                const float* __restrict__ gamma,
                                const float* __restrict__ beta,
                                float* __restrict__ aff, int cout) {
  int o = threadIdx.x;
  if (o < cout) {
    const float cnt = (float)NPTS * (float)KNN_K;
    float mu  = stats[o] / cnt;
    float var = stats[cout + o] / cnt - mu * mu;
    float a   = gamma[o] * rsqrtf(var + EPSV);
    aff[o]        = a;
    aff[cout + o] = beta[o] - mu * a;
  }
}

// ---------------------------------------------------------------------------
// EdgeConv pass B: recompute h (WMMA), norm affine, attention softmax,
// h @ att, distance-gated max -> x_out[N][COUT].
// Per-wave LDS buffer unioned: E_T (f16) first, then hn stored [k][COUT] (f32)
// so each mtile writes two contiguous b128 and Kv/Qv read contiguous float4.
// ---------------------------------------------------------------------------
template <int CIN, int COUT, int KPAD>
__global__ __launch_bounds__(256) void edge_apply_kernel(
    const float* __restrict__ xin, const int* __restrict__ knn_in,
    const float* __restrict__ vc,
    const float* __restrict__ Wg, const float* __restrict__ bg,
    const float* __restrict__ wkv, const float* __restrict__ wqv,
    const float* __restrict__ aff, float* __restrict__ xout) {
  constexpr int MT = COUT / 16, KT = KPAD / 32;
  __shared__ _Float16 sW[COUT * KPAD];
  __shared__ float    sB[COUT], sA[COUT], sSh[COUT], sWk[COUT], sWq[COUT];
  __shared__ float    sEH[8][16 * COUT];  // per wave: E_T(f16) then hn[k][COUT]
  __shared__ float    sKv[8][16], sQv[8][16], sSc[8][16];
  __shared__ float    sAtt[8][16 * 16];

  const int tid = threadIdx.x, lane = tid & 31, wv = tid >> 5;
  for (int t = tid; t < COUT * KPAD; t += 256) {
    int o = t / KPAD, kk = t % KPAD;
    sW[t] = (kk < 2 * CIN) ? (_Float16)Wg[o * 2 * CIN + kk] : (_Float16)0.f;
  }
  for (int t = tid; t < COUT; t += 256) {
    sB[t] = bg[t]; sA[t] = aff[t]; sSh[t] = aff[COUT + t];
    sWk[t] = wkv[t]; sWq[t] = wqv[t];
  }
  __syncthreads();

  v16h afr[MT][KT];
  {
    const int Mrow = lane & 15;
    const int kof  = (lane < 16) ? 0 : 8;
#pragma unroll
    for (int mt = 0; mt < MT; ++mt)
#pragma unroll
      for (int kt = 0; kt < KT; ++kt)
#pragma unroll
        for (int v = 0; v < 8; ++v) {
          int kk = kt * 32 + (v >> 2) * 16 + kof + 2 * (v & 3);
          afr[mt][kt][2 * v]     = sW[(mt * 16 + Mrow) * KPAD + kk];
          afr[mt][kt][2 * v + 1] = sW[(mt * 16 + Mrow) * KPAD + kk + 1];
        }
  }

  _Float16* e  = (_Float16*)sEH[wv];
  float*    hw = sEH[wv];                 // hn layout: [k][COUT]
  const int k    = lane & 15;
  const int half = lane >> 4;
  const int n0   = (blockIdx.x * 8 + wv) * PPW;

  for (int pi = 0; pi < PPW; ++pi) {
    const int n = n0 + pi;
    wave_lds_fence();                     // prior-point reads done before E rewrite
    const int j = knn_in[(size_t)n * KNN_K + k];
    build_eT<CIN, COUT, KPAD>(e, xin, n, j, k, half);
    wave_lds_fence();

    v16h bfr[KT];
    load_bfr<KPAD>(bfr, e, lane);
    wave_lds_fence();                     // B reads done before hn overwrites E

    // WMMA + bias + relu + norm affine -> hw[k][COUT], two b128 per mtile
#pragma unroll
    for (int mt = 0; mt < MT; ++mt) {
      v8f cf = {};
#pragma unroll
      for (int kt = 0; kt < KT; ++kt)
        cf = __builtin_amdgcn_wmma_f32_16x16x32_f16(false, afr[mt][kt], false, bfr[kt],
                                                    (short)0, cf, false, false);
      float tmp[8];
#pragma unroll
      for (int r = 0; r < 8; ++r) {
        int   M   = mt * 16 + r + half * 8;
        float val = cf[r] + sB[M];
        val       = val > 0.f ? val : 0.f;
        tmp[r]    = val * sA[M] + sSh[M];
      }
      float* dst = hw + k * COUT + mt * 16 + half * 8;
      ((float4*)dst)[0] = make_float4(tmp[0], tmp[1], tmp[2], tmp[3]);
      ((float4*)dst)[1] = make_float4(tmp[4], tmp[5], tmp[6], tmp[7]);
    }
    wave_lds_fence();

    // Kv / Qv (contiguous float4 rows) + distance gate for half 0
    {
      const float* hrow = hw + k * COUT;
      const float* wvec = half ? sWq : sWk;
      float acc = 0.f;
#pragma unroll
      for (int o = 0; o < COUT; o += 4) {
        float4 hv = *(const float4*)(hrow + o);
        float4 wf = *(const float4*)(wvec + o);
        acc += hv.x * wf.x + hv.y * wf.y + hv.z * wf.z + hv.w * wf.w;
      }
      if (half == 0) {
        sKv[wv][k] = acc;
        float dx  = vc[(size_t)j * 4 + 1] - vc[(size_t)n * 4 + 1];
        float dy  = vc[(size_t)j * 4 + 2] - vc[(size_t)n * 4 + 2];
        float dz  = vc[(size_t)j * 4 + 3] - vc[(size_t)n * 4 + 3];
        float dis = sqrtf(dx * dx + dy * dy + dz * dz);
        sSc[wv][k] = 2.f / (1.f + expf(dis));   // 2*sigmoid(-dis)
      } else {
        sQv[wv][k] = acc;
      }
    }
    wave_lds_fence();

    // att[:,kk,q] = softmax_kk(Kv[kk]*Qv[q]); lane q = k (lanes 0..15)
    if (half == 0) {
      const float qv = sQv[wv][k];
      float mx = -3.4e38f;
#pragma unroll
      for (int kk = 0; kk < 16; ++kk) mx = fmaxf(mx, sKv[wv][kk] * qv);
      float ssum = 0.f;
#pragma unroll
      for (int kk = 0; kk < 16; ++kk) {
        float ee = expf(sKv[wv][kk] * qv - mx);
        sAtt[wv][kk * 16 + k] = ee;
        ssum += ee;
      }
      float inv = 1.f / ssum;
#pragma unroll
      for (int kk = 0; kk < 16; ++kk) sAtt[wv][kk * 16 + k] *= inv;
    }
    wave_lds_fence();

    // out[o] = max_q scale[q] * sum_kk hw[kk][o]*att[kk][q]
    for (int o = lane; o < COUT; o += 32) {
      float hv[16];
#pragma unroll
      for (int kk = 0; kk < 16; ++kk) hv[kk] = hw[kk * COUT + o];
      float best = -3.4e38f;
#pragma unroll
      for (int q = 0; q < 16; ++q) {
        float acc = 0.f;
#pragma unroll
        for (int kk = 0; kk < 16; ++kk) acc += hv[kk] * sAtt[wv][kk * 16 + q];
        best = fmaxf(best, sSc[wv][q] * acc);
      }
      xout[(size_t)n * COUT + o] = best;
    }
  }
}

// ---------------------------------------------------------------------------
extern "C" void kernel_launch(void* const* d_in, const int* in_sizes, int n_in,
                              void* d_out, int out_size, void* d_ws, size_t ws_size,
                              hipStream_t stream) {
  (void)in_sizes; (void)n_in; (void)out_size; (void)ws_size;
  const float* pf  = (const float*)d_in[0];
  const float* vc  = (const float*)d_in[1];
  const float* W0  = (const float*)d_in[2];
  const float* b0  = (const float*)d_in[3];
  const float* g0  = (const float*)d_in[4];
  const float* be0 = (const float*)d_in[5];
  const float* wk0 = (const float*)d_in[6];
  const float* wq0 = (const float*)d_in[7];
  const float* W1  = (const float*)d_in[8];
  const float* b1  = (const float*)d_in[9];
  const float* g1  = (const float*)d_in[10];
  const float* be1 = (const float*)d_in[11];
  const float* wk1 = (const float*)d_in[12];
  const float* wq1 = (const float*)d_in[13];
  float* out = (float*)d_out;

  char*  ws     = (char*)d_ws;
  int*   knnb   = (int*)ws;                                        // 4 MiB
  float* x1     = (float*)(ws + (size_t)NPTS * KNN_K * 4);         // 8 MiB
  float* stats0 = (float*)(ws + (size_t)NPTS * KNN_K * 4 + (size_t)NPTS * 32 * 4);
  float* stats1 = stats0 + 64;
  float* aff0   = stats1 + 128;
  float* aff1   = aff0 + 64;

  hipMemsetAsync(stats0, 0, (64 + 128) * sizeof(float), stream);

  knn_kernel<<<BATCH * 128, 256, 0, stream>>>(vc, knnb);

  edge_stats_kernel<4, 32, 32><<<NPTS / (8 * PPW), 256, 0, stream>>>(pf, knnb, W0, b0, stats0);
  finalize_kernel<<<1, 64, 0, stream>>>(stats0, g0, be0, aff0, 32);
  edge_apply_kernel<4, 32, 32><<<NPTS / (8 * PPW), 256, 0, stream>>>(
      pf, knnb, vc, W0, b0, wk0, wq0, aff0, x1);

  edge_stats_kernel<32, 64, 64><<<NPTS / (8 * PPW), 256, 0, stream>>>(x1, knnb, W1, b1, stats1);
  finalize_kernel<<<1, 64, 0, stream>>>(stats1, g1, be1, aff1, 64);
  edge_apply_kernel<32, 64, 64><<<NPTS / (8 * PPW), 256, 0, stream>>>(
      x1, knnb, vc, W1, b1, wk1, wq1, aff1, out);

  hipMemcpyAsync(out + (size_t)NPTS * 64, vc, (size_t)NPTS * 4 * sizeof(float),
                 hipMemcpyDeviceToDevice, stream);
}
